// TopKGCNMultiTask_29300266893896
// MI455X (gfx1250) — matrix operand order repaired and checked by reading, not compiled, not run on hardware
//
#include <hip/hip_runtime.h>
#include <hip/hip_bf16.h>

typedef __attribute__((ext_vector_type(16))) __bf16 v16bf;
typedef __attribute__((ext_vector_type(8)))  float  v8f;

namespace {
constexpr int kB    = 8192;   // graphs
constexpr int kNPG  = 32;     // nodes per graph
constexpr int kKeep = 16;     // top-k
constexpr int kEPG  = 64;     // edges per graph
constexpr int kN    = kB * kNPG;
constexpr int kE    = kB * kEPG;
constexpr int kT    = 12;     // tasks
constexpr int kFIN  = 64;
constexpr int kG    = 128;    // hidden
constexpr int kH    = 64;
}

// ---------------------------------------------------------------------------
// WMMA fragment builders (CDNA5 ISA 7.12.2, 16-bit layouts, wave32)
// A: 16x32 (MxK). lanes 0-15: M=0..15; lanes 16-31: same M, K-half shifted by 8.
//    VGPR i (i<4): K = 2i,2i+1 (+8 if upper half); VGPR i>=4: K += 16.
// ---------------------------------------------------------------------------
__device__ __forceinline__ v16bf load_a_frag_lds(const __bf16* s, int row, int ld,
                                                 int k0, int half) {
  v16bf a;
#pragma unroll
  for (int i = 0; i < 8; ++i) {
    int kbase = 2 * i + 8 * half + ((i >= 4) ? 8 : 0);
    a[2 * i]     = s[row * ld + k0 + kbase];
    a[2 * i + 1] = s[row * ld + k0 + kbase + 1];
  }
  return a;
}

// B: 32x16 (KxN), row-striped across lanes. lane -> N = lane&15.
//    VGPR i (i<4): K = 2i,2i+1 (+8 if upper half); VGPR i>=4: K += 16 (same as A).
__device__ __forceinline__ v16bf load_b_frag_glb(const float* __restrict__ W, int col,
                                                 int ldn, int k0, int half) {
  v16bf b;
#pragma unroll
  for (int i = 0; i < 8; ++i) {
    int k = k0 + 2 * i + 8 * half + ((i >= 4) ? 8 : 0);
    b[2 * i]     = (__bf16)W[k * ldn + col];
    b[2 * i + 1] = (__bf16)W[(k + 1) * ldn + col];
  }
  return b;
}

// ---------------------------------------------------------------------------
// Fused GCN layer: h_out = relu( agg + h*dinv^2 + b ),  h = x @ W
// One workgroup (8 wave32) per graph. WMMA bf16 -> f32.
// ---------------------------------------------------------------------------
template <int FIN, bool IN_BF16>
__global__ __launch_bounds__(256) void gcn_layer_kernel(
    const void* __restrict__ xin_v,   // [N, FIN] f32 (layer1) or bf16 (layer2)
    const float* __restrict__ W,      // [FIN, 128] f32 (L2-resident)
    const float* __restrict__ bias,   // [128]
    const int* __restrict__ src,      // [E] global node ids
    const int* __restrict__ dst,      // [E]
    __bf16* __restrict__ hout)        // [N, 128] bf16 (may alias xin for layer2)
{
  const int g    = blockIdx.x;
  const int tid  = threadIdx.x;
  const int wave = tid >> 5;
  const int lane = tid & 31;
  const int half = lane >> 4;
  const int r    = lane & 15;

  __shared__ __bf16 sX[kNPG * FIN];
  __shared__ float  sH[kNPG * kG];
  __shared__ float  sAgg[kNPG * kG];
  __shared__ float  sDinv[kNPG];
  __shared__ float  sDeg[kNPG];
  __shared__ int    sSrc[kEPG];
  __shared__ int    sDst[kEPG];

  // phase 0: stage inputs
  if (tid < kNPG) sDeg[tid] = 0.0f;
  if (IN_BF16) {
    const __bf16* xin = (const __bf16*)xin_v;
    for (int i = tid; i < kNPG * FIN; i += 256)
      sX[i] = xin[(size_t)g * kNPG * FIN + i];
  } else {
    const float* xin = (const float*)xin_v;
    for (int i = tid; i < kNPG * FIN; i += 256)
      sX[i] = (__bf16)xin[(size_t)g * kNPG * FIN + i];
    if (tid == 0 && g + 1 < kB)  // hint next tile into GL2
      __builtin_prefetch(xin + (size_t)(g + 1) * kNPG * FIN, 0, 1);
  }
  if (tid < kEPG) {
    sSrc[tid] = src[g * kEPG + tid] - g * kNPG;
    sDst[tid] = dst[g * kEPG + tid] - g * kNPG;
  }
  __syncthreads();

  // phase 1: degrees + WMMA GEMM  h = x @ W  (per-wave 16-col tile, both row tiles)
  if (tid < kEPG) atomicAdd(&sDeg[sDst[tid]], 1.0f);

  v8f acc0 = {};
  v8f acc1 = {};
#pragma unroll
  for (int ks = 0; ks < FIN / 32; ++ks) {
    const int k0 = ks * 32;
    v16bf bfrag = load_b_frag_glb(W, wave * 16 + r, kG, k0, half);
    v16bf a0 = load_a_frag_lds(sX, r, FIN, k0, half);
    v16bf a1 = load_a_frag_lds(sX, 16 + r, FIN, k0, half);
    acc0 = __builtin_amdgcn_wmma_f32_16x16x32_bf16(false, a0, false, bfrag,
                                                   (short)0, acc0, false, false);
    acc1 = __builtin_amdgcn_wmma_f32_16x16x32_bf16(false, a1, false, bfrag,
                                                   (short)0, acc1, false, false);
  }
  // C/D layout: VGPR i -> M = i + 8*half, N = lane&15
#pragma unroll
  for (int i = 0; i < 8; ++i) {
    sH[(i + 8 * half) * kG + wave * 16 + r]        = acc0[i];
    sH[(16 + i + 8 * half) * kG + wave * 16 + r]   = acc1[i];
  }
  __syncthreads();

  // phase 2: dinv + zero agg
  if (tid < kNPG) sDinv[tid] = rsqrtf(sDeg[tid] + 1.0f);
  for (int i = tid; i < kNPG * kG; i += 256) sAgg[i] = 0.0f;
  __syncthreads();

  // phase 3: edge aggregation (LDS float atomics)
  {
    const int f    = tid & (kG - 1);  // 0..127
    const int esub = tid >> 7;        // 0..1
    for (int e = esub; e < kEPG; e += 2) {
      const int s = sSrc[e], d = sDst[e];
      const float coef = sDinv[s] * sDinv[d];
      atomicAdd(&sAgg[d * kG + f], sH[s * kG + f] * coef);
    }
  }
  __syncthreads();

  // phase 4: out = relu(agg + h*dinv^2 + b) -> bf16
  for (int i = tid; i < kNPG * kG; i += 256) {
    const int n = i >> 7;
    const int f = i & (kG - 1);
    const float di = sDinv[n];
    float v = sAgg[i] + sH[i] * di * di + bias[f];
    v = fmaxf(v, 0.0f);
    hout[(size_t)(g * kNPG + n) * kG + f] = (__bf16)v;
  }
}

// ---------------------------------------------------------------------------
// Per-task precompute: q_t = tW_t @ lW_t  [128], c_t = 16*(tb_t . lW_t) + lb_t,
// pn_t = ||p_t||.  One block per task.
// ---------------------------------------------------------------------------
__global__ __launch_bounds__(128) void task_precompute_kernel(
    const float* __restrict__ tconv_W,  // [T,128,64]
    const float* __restrict__ tconv_b,  // [T,64]
    const float* __restrict__ lin_W,    // [T,64,1]
    const float* __restrict__ lin_b,    // [T,1]
    const float* __restrict__ pool_p,   // [T,128]
    float* __restrict__ q,              // [T,128]
    float* __restrict__ cpn)            // [2T]: c then pnorm
{
  const int t = blockIdx.x;
  const int i = threadIdx.x;  // 0..127
  const float* lw = lin_W + t * kH;

  float acc = 0.0f;
  const float* tw = tconv_W + ((size_t)t * kG + i) * kH;
  for (int h = 0; h < kH; ++h) acc += tw[h] * lw[h];
  q[t * kG + i] = acc;

  __shared__ float red[128];
  red[i] = (i < kH) ? tconv_b[t * kH + i] * lw[i] : 0.0f;
  __syncthreads();
  for (int s = 64; s > 0; s >>= 1) {
    if (i < s) red[i] += red[i + s];
    __syncthreads();
  }
  if (i == 0) cpn[t] = (float)kKeep * red[0] + lin_b[t];
  __syncthreads();
  const float pv = pool_p[t * kG + i];
  red[i] = pv * pv;
  __syncthreads();
  for (int s = 64; s > 0; s >>= 1) {
    if (i < s) red[i] += red[i + s];
    __syncthreads();
  }
  if (i == 0) cpn[kT + t] = sqrtf(red[0]);
}

// ---------------------------------------------------------------------------
// Collapsed TopK-pool multi-task head. One workgroup per graph, loops 12 tasks.
// out[g*T + t] = sigmoid( sum_j wgt_j * gate_j * (h2[perm_j] . q_t) + c_t )
// ---------------------------------------------------------------------------
__global__ __launch_bounds__(256) void topk_task_kernel(
    const __bf16* __restrict__ h2,     // [N,128] bf16
    const float* __restrict__ pool_p,  // [T,128]
    const int* __restrict__ src,
    const int* __restrict__ dst,
    const float* __restrict__ q,       // [T,128]
    const float* __restrict__ cpn,     // [2T]
    float* __restrict__ out)           // [B*T]
{
  const int g = blockIdx.x;
  const int tid = threadIdx.x;

  __shared__ float sH[kNPG * kG];
  __shared__ int   sSrc[kEPG], sDst[kEPG];
  __shared__ float sP[kG], sQ[kG];
  __shared__ float sPart[256];
  __shared__ float sScore[kNPG];
  __shared__ int   sNew[kNPG];     // rank if kept else -1
  __shared__ int   sPerm[kKeep];
  __shared__ float sGate[kKeep];
  __shared__ float sDegP[kKeep];
  __shared__ float sDinvP[kKeep];
  __shared__ float sWgt[kKeep];
  __shared__ float sRes;

  for (int i = tid; i < kNPG * kG; i += 256)
    sH[i] = (float)h2[(size_t)g * kNPG * kG + i];
  if (tid < kEPG) {
    sSrc[tid] = src[g * kEPG + tid] - g * kNPG;
    sDst[tid] = dst[g * kEPG + tid] - g * kNPG;
  }
  __syncthreads();

  for (int t = 0; t < kT; ++t) {
    if (tid < kG) {
      sP[tid] = pool_p[t * kG + tid];
      sQ[tid] = q[t * kG + tid];
    }
    __syncthreads();

    // scores: node = tid>>3, 8 partials x 16 features
    {
      const int n = tid >> 3, c = tid & 7;
      float acc = 0.0f;
      const float* hh = sH + n * kG + c * 16;
      const float* pp = sP + c * 16;
#pragma unroll
      for (int f = 0; f < 16; ++f) acc += hh[f] * pp[f];
      sPart[tid] = acc;
    }
    __syncthreads();
    if (tid < kNPG) {
      float s = 0.0f;
#pragma unroll
      for (int j = 0; j < 8; ++j) s += sPart[tid * 8 + j];
      sScore[tid] = s / cpn[kT + t];
    }
    __syncthreads();

    // stable top-16 via rank (matches jax.lax.top_k ordering + ties)
    if (tid < kNPG) {
      const float mine = sScore[tid];
      int rank = 0;
      for (int j = 0; j < kNPG; ++j) {
        const float o = sScore[j];
        rank += (o > mine) || (o == mine && j < tid);
      }
      sNew[tid] = (rank < kKeep) ? rank : -1;
      if (rank < kKeep) {
        sPerm[rank] = tid;
        sGate[rank] = tanhf(mine);
      }
    }
    if (tid < kKeep) sDegP[tid] = 0.0f;
    __syncthreads();

    // pooled-graph degrees over surviving edges
    if (tid < kEPG) {
      const int ns = sNew[sSrc[tid]], nd = sNew[sDst[tid]];
      if (ns >= 0 && nd >= 0) atomicAdd(&sDegP[nd], 1.0f);
    }
    __syncthreads();
    if (tid < kKeep) {
      const float di = rsqrtf(sDegP[tid] + 1.0f);
      sDinvP[tid] = di;
      sWgt[tid]   = di * di;  // self-loop term
    }
    if (tid == 0) sRes = 0.0f;
    __syncthreads();
    if (tid < kEPG) {
      const int ns = sNew[sSrc[tid]], nd = sNew[sDst[tid]];
      if (ns >= 0 && nd >= 0) atomicAdd(&sWgt[ns], sDinvP[ns] * sDinvP[nd]);
    }
    __syncthreads();

    // z = sum_j wgt_j*gate_j*(h2[perm_j].q) ; node = tid>>4, 16 partials x 8 feats
    {
      const int j = tid >> 4, c = tid & 15;
      const float* hh = sH + sPerm[j] * kG + c * 8;
      const float* qq = sQ + c * 8;
      float acc = 0.0f;
#pragma unroll
      for (int f = 0; f < 8; ++f) acc += hh[f] * qq[f];
      atomicAdd(&sRes, acc * sWgt[j] * sGate[j]);
    }
    __syncthreads();
    if (tid == 0) {
      const float z = sRes + cpn[t];
      out[(size_t)g * kT + t] = 1.0f / (1.0f + __expf(-z));
    }
    __syncthreads();
  }
}

// ---------------------------------------------------------------------------
extern "C" void kernel_launch(void* const* d_in, const int* in_sizes, int n_in,
                              void* d_out, int out_size, void* d_ws, size_t ws_size,
                              hipStream_t stream) {
  (void)in_sizes; (void)n_in; (void)out_size; (void)ws_size;

  const float* x       = (const float*)d_in[0];
  const float* gc1_W   = (const float*)d_in[1];
  const float* gc1_b   = (const float*)d_in[2];
  const float* gc2_W   = (const float*)d_in[3];
  const float* gc2_b   = (const float*)d_in[4];
  const float* pool_p  = (const float*)d_in[5];
  const float* tconv_W = (const float*)d_in[6];
  const float* tconv_b = (const float*)d_in[7];
  const float* lin_W   = (const float*)d_in[8];
  const float* lin_b   = (const float*)d_in[9];
  const int*   eidx    = (const int*)d_in[10];
  const int* src = eidx;
  const int* dst = eidx + kE;

  // workspace: h (bf16, N x 128) | q (f32, T x 128) | cpn (f32, 2T)
  __bf16* h  = (__bf16*)d_ws;
  float*  q  = (float*)((char*)d_ws + (size_t)kN * kG * sizeof(__bf16));
  float* cpn = q + kT * kG;

  gcn_layer_kernel<kFIN, false><<<kB, 256, 0, stream>>>(x, gc1_W, gc1_b, src, dst, h);
  gcn_layer_kernel<kG,   true ><<<kB, 256, 0, stream>>>(h, gc2_W, gc2_b, src, dst, h);
  task_precompute_kernel<<<kT, 128, 0, stream>>>(tconv_W, tconv_b, lin_W, lin_b,
                                                 pool_p, q, cpn);
  topk_task_kernel<<<kB, 256, 0, stream>>>(h, pool_p, src, dst, q, cpn, (float*)d_out);
}
